// GINE0_14285061227137
// MI455X (gfx1250) — matrix operand order repaired and checked by reading, not compiled, not run on hardware
//
#include <hip/hip_runtime.h>
#include <hip/hip_bf16.h>
#include <math.h>

typedef __attribute__((ext_vector_type(16))) __bf16 v16bf;
typedef __attribute__((ext_vector_type(8)))  float  v8f;

#define N_NODES  50000
#define N_EDGES  500000
#define EDGE_DIM 16
#define DIM      128
#define N_LAYERS 3
#define N_GRAPHS 512
#define N_CLASSES 10
#define BN_EPS   1e-5f

#define EDGE_GRID 2048   // persistent blocks for edge pipeline (7813 tiles)
#define NODE_GRID 782    // node tiles = 782, one each

// ---------------------------------------------------------------------------
// WMMA helpers (CDNA5 wave32, V_WMMA_F32_16X16X32_BF16)
// ---------------------------------------------------------------------------

__device__ __forceinline__ v8f wmma_bf16(v16bf a, v16bf b, v8f c) {
  // (neg_a, A, neg_b, B, c_mod, C, reuse_a, reuse_b)
  return __builtin_amdgcn_wmma_f32_16x16x32_bf16(false, a, false, b,
                                                 (short)0, c, false, false);
}

// A-matrix fragment 16x32 bf16 from LDS (row-major, ldu elements per row).
// ISA layout: lanes 0-15 -> M=0..15; lanes 16-31 -> M=0..15 (second K half).
// VGPR pair p holds K = 2p (+8 for p>=4) (+8 for lanes 16-31) .. +1
__device__ __forceinline__ v16bf load_a_frag(const __bf16* base, int ldu, int lane) {
  const int m    = lane & 15;
  const int half = lane >> 4;
  const uint32_t* row = reinterpret_cast<const uint32_t*>(base + m * ldu);
  union { v16bf v; uint32_t u[8]; } A;
#pragma unroll
  for (int p = 0; p < 8; ++p) {
    int k = 2 * p + ((p >= 4) ? 8 : 0) + 8 * half;   // even
    A.u[p] = row[k >> 1];
  }
  return A.v;
}

// B-matrix fragment 32x16 bf16 from row-major [K][ldw] global weights.
// element e: K = k0 + e (+8 for e>=8) (+8 for lanes 16-31); n = n0 + lane%16.
__device__ __forceinline__ v16bf load_b_frag(const __bf16* __restrict__ W,
                                             int ldw, int k0, int n0, int lane) {
  const int n  = n0 + (lane & 15);
  const int kb = k0 + ((lane >> 4) ? 8 : 0);
  union { v16bf v; __bf16 h[16]; } B;
#pragma unroll
  for (int e = 0; e < 16; ++e) B.h[e] = W[(kb + e + ((e >= 8) ? 8 : 0)) * ldw + n];
  return B.v;
}

// ---------------------------------------------------------------------------
// Weight prep: f32 -> bf16 (ee_w1 zero-padded K 16->32)
// ---------------------------------------------------------------------------

__global__ void cvt_bf16_kernel(const float* __restrict__ src,
                                __bf16* __restrict__ dst, int n) {
  int i = blockIdx.x * blockDim.x + threadIdx.x;
  if (i < n) dst[i] = (__bf16)src[i];
}

__global__ void pad_w1_kernel(const float* __restrict__ src,  // [3][16][128]
                              __bf16* __restrict__ dst) {     // [3][32][128]
  int i = blockIdx.x * blockDim.x + threadIdx.x;
  if (i >= N_LAYERS * 32 * DIM) return;
  int l = i / (32 * DIM);
  int r = i - l * 32 * DIM;
  int k = r / DIM, n = r - k * DIM;
  float v = (k < EDGE_DIM) ? src[(l * EDGE_DIM + k) * DIM + n] : 0.f;
  dst[i] = (__bf16)v;
}

__global__ void zero_kernel(float* __restrict__ p, int n) {
  int i = blockIdx.x * blockDim.x + threadIdx.x;
  int stride = gridDim.x * blockDim.x;
  for (; i < n; i += stride) p[i] = 0.f;
}

// ---------------------------------------------------------------------------
// Fused edge pipeline: edge-encoder (2 GEMMs + bias + relu + BN) -> message
// relu(x[src]+e) -> atomic scatter-add into agg[dst].
// Persistent: block = 256 thr = 8 waves; grid-strides over 64-edge tiles,
// B fragments stay resident in VGPRs across tiles.
// ---------------------------------------------------------------------------

__global__ __launch_bounds__(256) void edge_kernel(
    const float* __restrict__ edge_attr,   // [E][16]
    const int*   __restrict__ ei,          // [2][E] (src, dst)
    const float* __restrict__ xin,         // [N][128]
    const __bf16* __restrict__ W1,         // [32][128] padded bf16
    const float* __restrict__ B1,          // [128]
    const __bf16* __restrict__ W2,         // [128][128] bf16
    const float* __restrict__ B2,
    const float* __restrict__ G, const float* __restrict__ Bb,
    const float* __restrict__ RM, const float* __restrict__ RV,
    float* __restrict__ agg, int E) {
  __shared__ __align__(16) __bf16 sA[64 * 32];    // edge_attr tile, K padded
  __shared__ __align__(16) __bf16 sE[64 * 128];   // hidden after GEMM1

  const int tid  = threadIdx.x;
  const int wave = tid >> 5;
  const int lane = tid & 31;
  const int n0   = wave * 16;

  // Resident B fragments for this wave's column slice (live across tiles).
  v16bf b1f = load_b_frag(W1, DIM, 0, n0, lane);
  v16bf b2f[4];
#pragma unroll
  for (int kt = 0; kt < 4; ++kt) b2f[kt] = load_b_frag(W2, DIM, kt * 32, n0, lane);

  const int n    = n0 + (lane & 15);
  const int half = lane >> 4;
  const float bias1 = B1[n];
  const float bias2 = B2[n];
  const float scale = G[n] * rsqrtf(RV[n] + BN_EPS);
  const float shift = Bb[n] - RM[n] * scale;

  const int num_tiles = (E + 63) >> 6;

  for (int tile = blockIdx.x; tile < num_tiles; tile += gridDim.x) {
    const int e0 = tile * 64;

    // Prefetch next tile's edge_attr (4 KB) + edge indices into cache.
    const int nt = tile + gridDim.x;
    if (nt < num_tiles) {
      const int ne0 = nt * 64;
      if (tid < 32)
        __builtin_prefetch(edge_attr + (size_t)ne0 * EDGE_DIM + tid * 32, 0, 1);
      if (tid == 32) __builtin_prefetch(ei + ne0, 0, 1);
      if (tid == 33) __builtin_prefetch(ei + E + ne0, 0, 1);
    }

    // Stage edge_attr tile (zero-pad K 16..31, zero-pad past-end edges).
    for (int i = tid; i < 64 * 32; i += 256) {
      int r = i >> 5, k = i & 31;
      int e = e0 + r;
      float v = (k < EDGE_DIM && e < E) ? edge_attr[e * EDGE_DIM + k] : 0.f;
      sA[i] = (__bf16)v;
    }
    __syncthreads();

    // GEMM1: [64,32]@[32,16-slice] -> relu -> sE (bf16)
#pragma unroll
    for (int mt = 0; mt < 4; ++mt) {
      v16bf a = load_a_frag(&sA[(mt * 16) * 32], 32, lane);
      v8f c;
#pragma unroll
      for (int r = 0; r < 8; ++r) c[r] = bias1;
      c = wmma_bf16(a, b1f, c);
#pragma unroll
      for (int r = 0; r < 8; ++r) {
        int row = mt * 16 + r + 8 * half;
        sE[row * DIM + n] = (__bf16)fmaxf(c[r], 0.f);
      }
    }
    __syncthreads();

    // GEMM2: [64,128]@[128,16-slice] -> relu -> BN -> message -> scatter-add
#pragma unroll
    for (int mt = 0; mt < 4; ++mt) {
      v8f c;
#pragma unroll
      for (int r = 0; r < 8; ++r) c[r] = bias2;
#pragma unroll
      for (int kt = 0; kt < 4; ++kt) {
        v16bf a = load_a_frag(&sE[(mt * 16) * DIM + kt * 32], DIM, lane);
        c = wmma_bf16(a, b2f[kt], c);
      }
#pragma unroll
      for (int r = 0; r < 8; ++r) {
        int row = mt * 16 + r + 8 * half;
        int e = e0 + row;
        if (e < E) {
          float ev  = fmaxf(c[r], 0.f) * scale + shift;       // relu then BN
          int s = ei[e], d = ei[E + e];
          float msg = fmaxf(xin[s * DIM + n] + ev, 0.f);
          atomicAdd(&agg[d * DIM + n], msg);
        }
      }
    }
    // Next iteration's staging writes only sA, which is consumed strictly
    // between the two barriers above -> no extra barrier needed here.
  }
}

// ---------------------------------------------------------------------------
// Node MLP: h = x + agg; relu(h@W1+b1); relu(@W2+b2); BN -> xout
// ---------------------------------------------------------------------------

__global__ __launch_bounds__(256) void node_kernel(
    const float* __restrict__ xin, const float* __restrict__ agg,
    const __bf16* __restrict__ W1, const float* __restrict__ B1,
    const __bf16* __restrict__ W2, const float* __restrict__ B2,
    const float* __restrict__ G, const float* __restrict__ Bb,
    const float* __restrict__ RM, const float* __restrict__ RV,
    float* __restrict__ xout, int Nn) {
  __shared__ __align__(16) __bf16 sX[64 * 128];
  __shared__ __align__(16) __bf16 sH[64 * 128];

  const int tid  = threadIdx.x;
  const int wave = tid >> 5;
  const int lane = tid & 31;
  const int n0   = wave * 16;

  v16bf b1f[4], b2f[4];
#pragma unroll
  for (int kt = 0; kt < 4; ++kt) {
    b1f[kt] = load_b_frag(W1, DIM, kt * 32, n0, lane);
    b2f[kt] = load_b_frag(W2, DIM, kt * 32, n0, lane);
  }

  const int n    = n0 + (lane & 15);
  const int half = lane >> 4;
  const float bias1 = B1[n];
  const float bias2 = B2[n];
  const float scale = G[n] * rsqrtf(RV[n] + BN_EPS);
  const float shift = Bb[n] - RM[n] * scale;

  const int num_tiles = (Nn + 63) >> 6;

  for (int tile = blockIdx.x; tile < num_tiles; tile += gridDim.x) {
    const int v0 = tile * 64;

    for (int i = tid; i < 64 * DIM; i += 256) {
      int r = i >> 7, d = i & 127;
      int node = v0 + r;
      float v = (node < Nn) ? (xin[node * DIM + d] + agg[node * DIM + d]) : 0.f;
      sX[i] = (__bf16)v;
    }
    __syncthreads();

#pragma unroll
    for (int mt = 0; mt < 4; ++mt) {
      v8f c;
#pragma unroll
      for (int r = 0; r < 8; ++r) c[r] = bias1;
#pragma unroll
      for (int kt = 0; kt < 4; ++kt) {
        v16bf a = load_a_frag(&sX[(mt * 16) * DIM + kt * 32], DIM, lane);
        c = wmma_bf16(a, b1f[kt], c);
      }
#pragma unroll
      for (int r = 0; r < 8; ++r) {
        int row = mt * 16 + r + 8 * half;
        sH[row * DIM + n] = (__bf16)fmaxf(c[r], 0.f);
      }
    }
    __syncthreads();

#pragma unroll
    for (int mt = 0; mt < 4; ++mt) {
      v8f c;
#pragma unroll
      for (int r = 0; r < 8; ++r) c[r] = bias2;
#pragma unroll
      for (int kt = 0; kt < 4; ++kt) {
        v16bf a = load_a_frag(&sH[(mt * 16) * DIM + kt * 32], DIM, lane);
        c = wmma_bf16(a, b2f[kt], c);
      }
#pragma unroll
      for (int r = 0; r < 8; ++r) {
        int row = mt * 16 + r + 8 * half;
        int node = v0 + row;
        if (node < Nn)
          xout[node * DIM + n] = fmaxf(c[r], 0.f) * scale + shift;
      }
    }
  }
}

// ---------------------------------------------------------------------------
// Readout: mean pool per graph, then relu(@lin1+b) @ lin2 + b, log_softmax
// ---------------------------------------------------------------------------

__global__ void pool_kernel(const float* __restrict__ x,
                            const int* __restrict__ batch,
                            float* __restrict__ sums, float* __restrict__ cnts) {
  int i = blockIdx.x * blockDim.x + threadIdx.x;
  int stride = gridDim.x * blockDim.x;
  for (; i < N_NODES * DIM; i += stride) {
    int node = i >> 7, d = i & 127;
    int g = batch[node];
    atomicAdd(&sums[g * DIM + d], x[i]);
    if (d == 0) atomicAdd(&cnts[g], 1.f);
  }
}

__global__ __launch_bounds__(128) void head_kernel(
    const float* __restrict__ sums, const float* __restrict__ cnts,
    const float* __restrict__ w1, const float* __restrict__ b1,
    const float* __restrict__ w2, const float* __restrict__ b2,
    float* __restrict__ out) {
  __shared__ float p[DIM];
  __shared__ float h[DIM];
  __shared__ float lg[N_CLASSES];
  const int g = blockIdx.x, t = threadIdx.x;

  float c = fmaxf(cnts[g], 1.f);
  p[t] = sums[g * DIM + t] / c;
  __syncthreads();

  float acc = b1[t];
#pragma unroll 8
  for (int k = 0; k < DIM; ++k) acc += p[k] * w1[k * DIM + t];
  h[t] = fmaxf(acc, 0.f);
  __syncthreads();

  if (t < N_CLASSES) {
    float a = b2[t];
#pragma unroll 8
    for (int k = 0; k < DIM; ++k) a += h[k] * w2[k * N_CLASSES + t];
    lg[t] = a;
  }
  __syncthreads();

  if (t == 0) {
    float mx = lg[0];
#pragma unroll
    for (int i2 = 1; i2 < N_CLASSES; ++i2) mx = fmaxf(mx, lg[i2]);
    float s = 0.f;
#pragma unroll
    for (int i2 = 0; i2 < N_CLASSES; ++i2) s += __expf(lg[i2] - mx);
    float lse = mx + __logf(s);
#pragma unroll
    for (int i2 = 0; i2 < N_CLASSES; ++i2) out[g * N_CLASSES + i2] = lg[i2] - lse;
  }
}

// ---------------------------------------------------------------------------

extern "C" void kernel_launch(void* const* d_in, const int* in_sizes, int n_in,
                              void* d_out, int out_size, void* d_ws, size_t ws_size,
                              hipStream_t stream) {
  (void)in_sizes; (void)n_in; (void)out_size; (void)ws_size;
  const float* x          = (const float*)d_in[0];
  const float* edge_attr  = (const float*)d_in[1];
  const int*   edge_index = (const int*)d_in[2];
  const int*   batch      = (const int*)d_in[3];
  const float* ee_w1  = (const float*)d_in[4];
  const float* ee_b1  = (const float*)d_in[5];
  const float* ee_w2  = (const float*)d_in[6];
  const float* ee_b2  = (const float*)d_in[7];
  const float* ee_g   = (const float*)d_in[8];
  const float* ee_bb  = (const float*)d_in[9];
  const float* ee_rm  = (const float*)d_in[10];
  const float* ee_rv  = (const float*)d_in[11];
  const float* mw1    = (const float*)d_in[12];
  const float* mb1    = (const float*)d_in[13];
  const float* mw2    = (const float*)d_in[14];
  const float* mb2    = (const float*)d_in[15];
  const float* m_g    = (const float*)d_in[16];
  const float* m_bb   = (const float*)d_in[17];
  const float* m_rm   = (const float*)d_in[18];
  const float* m_rv   = (const float*)d_in[19];
  const float* lin1_w = (const float*)d_in[20];
  const float* lin1_b = (const float*)d_in[21];
  const float* lin2_w = (const float*)d_in[22];
  const float* lin2_b = (const float*)d_in[23];

  // workspace carve-up
  const size_t NF = (size_t)N_NODES * DIM;
  float* agg  = (float*)d_ws;
  float* xa   = agg + NF;
  float* xb   = xa + NF;
  float* sums = xb + NF;
  float* cnts = sums + (size_t)N_GRAPHS * DIM;
  __bf16* w      = (__bf16*)(cnts + N_GRAPHS);
  __bf16* ee_w1b = w;                                    // 3*32*128
  __bf16* ee_w2b = ee_w1b + (size_t)N_LAYERS * 32 * DIM; // 3*128*128
  __bf16* mw1b   = ee_w2b + (size_t)N_LAYERS * DIM * DIM;
  __bf16* mw2b   = mw1b   + (size_t)N_LAYERS * DIM * DIM;

  // weight conversion (deterministic each call)
  {
    int n1 = N_LAYERS * 32 * DIM;
    pad_w1_kernel<<<(n1 + 255) / 256, 256, 0, stream>>>(ee_w1, ee_w1b);
    int n2 = N_LAYERS * DIM * DIM;
    cvt_bf16_kernel<<<(n2 + 255) / 256, 256, 0, stream>>>(ee_w2, ee_w2b, n2);
    cvt_bf16_kernel<<<(n2 + 255) / 256, 256, 0, stream>>>(mw1, mw1b, n2);
    cvt_bf16_kernel<<<(n2 + 255) / 256, 256, 0, stream>>>(mw2, mw2b, n2);
  }

  const float* xin = x;
  float* bufs[2] = {xa, xb};

  for (int l = 0; l < N_LAYERS; ++l) {
    zero_kernel<<<4096, 256, 0, stream>>>(agg, (int)NF);
    edge_kernel<<<EDGE_GRID, 256, 0, stream>>>(
        edge_attr, edge_index, xin,
        ee_w1b + (size_t)l * 32 * DIM, ee_b1 + l * DIM,
        ee_w2b + (size_t)l * DIM * DIM, ee_b2 + l * DIM,
        ee_g + l * DIM, ee_bb + l * DIM, ee_rm + l * DIM, ee_rv + l * DIM,
        agg, N_EDGES);
    node_kernel<<<NODE_GRID, 256, 0, stream>>>(
        xin, agg,
        mw1b + (size_t)l * DIM * DIM, mb1 + l * DIM,
        mw2b + (size_t)l * DIM * DIM, mb2 + l * DIM,
        m_g + l * DIM, m_bb + l * DIM, m_rm + l * DIM, m_rv + l * DIM,
        bufs[l & 1], N_NODES);
    xin = bufs[l & 1];
  }

  zero_kernel<<<64, 256, 0, stream>>>(sums, N_GRAPHS * DIM + N_GRAPHS);
  pool_kernel<<<4096, 256, 0, stream>>>(xin, batch, sums, cnts);
  head_kernel<<<N_GRAPHS, DIM, 0, stream>>>(sums, cnts, lin1_w, lin1_b,
                                            lin2_w, lin2_b, (float*)d_out);
}
